// TransformerEncoder_27702539059690
// MI455X (gfx1250) — matrix operand order repaired and checked by reading, not compiled, not run on hardware
//
#include <hip/hip_runtime.h>
#include <hip/hip_bf16.h>
#include <stdint.h>

// ---------------------------------------------------------------------------
// Sizes (fixed by the reference)
// ---------------------------------------------------------------------------
#define E_   512
#define H_   8
#define DH_  64
#define S_   16
#define B_   128
#define MAXS 64
#define IN_  256
#define OUT_ 128
#define NL_  2
#define MEM_ (E_ * MAXS + 1)          // 32769
#define ROWS (B_ * S_)                // 2048 rows of state

typedef __attribute__((ext_vector_type(16))) __bf16 v16bf;
typedef __attribute__((ext_vector_type(8)))  __bf16 v8bf;
typedef __attribute__((ext_vector_type(8)))  float  v8f;

// A-fragment (16x32 bf16): lane m    -> K 0..15
//                          lane m+16 -> K 16..31
__device__ __forceinline__ v16bf load_a_frag(const __bf16* arow, int k0, int half) {
  v8bf lo = *(const v8bf*)(arow + k0     + half * 16);
  v8bf hi = *(const v8bf*)(arow + k0 + 8 + half * 16);
  return __builtin_shufflevector(lo, hi,
      0, 1, 2, 3, 4, 5, 6, 7, 8, 9, 10, 11, 12, 13, 14, 15);
}

// ---------------------------------------------------------------------------
// Generic bf16 WMMA GEMM: C[M,N] = A[M,K] * W[N,K]^T + bias (+bf16 mirror).
// 256 threads = 8 waves (2 M x 4 N), each wave a 32x32 tile (2x2 WMMA tiles,
// 4 independent accumulator chains).  Block tile 64x128.
// Requires M % 64 == 0, N % 128 == 0, K % 32 == 0.
// ---------------------------------------------------------------------------
__global__ __launch_bounds__(256) void gemm_bf16_wmma(
    const __bf16* __restrict__ A, int lda,
    const __bf16* __restrict__ W, int ldw,
    const float*  __restrict__ bias,
    float*        __restrict__ C, int ldc,
    __bf16*       __restrict__ Cb, int ldcb,
    int K)
{
  const int lane = threadIdx.x & 31;
  const int wave = threadIdx.x >> 5;          // 0..7
  const int wm   = wave >> 2;                 // 0..1
  const int wn   = wave & 3;                  // 0..3
  const int m0   = blockIdx.y * 64 + wm * 32;
  const int n0   = blockIdx.x * 128 + wn * 32;
  const int half = lane >> 4;
  const int l15  = lane & 15;

  const __bf16* arow0 = A + (size_t)(m0 + l15) * lda;
  const __bf16* arow1 = arow0 + (size_t)16 * lda;
  const __bf16* wrow0 = W + (size_t)(n0 + l15) * ldw;
  const __bf16* wrow1 = wrow0 + (size_t)16 * ldw;

  v8f acc00 = {}, acc01 = {}, acc10 = {}, acc11 = {};
  for (int k0 = 0; k0 < K; k0 += 32) {
    __builtin_prefetch(wrow0 + k0 + 64, 0, 1);
    v16bf a0 = load_a_frag(arow0, k0, half);
    v16bf a1 = load_a_frag(arow1, k0, half);
    // B-fragment (32x16, col n = W row n): lane n -> K 0..15, lane n+16 -> 16..31
    v16bf b0 = *(const v16bf*)(wrow0 + k0 + half * 16);
    v16bf b1 = *(const v16bf*)(wrow1 + k0 + half * 16);
    acc00 = __builtin_amdgcn_wmma_f32_16x16x32_bf16(false, a0, false, b0, (short)0, acc00, false, false);
    acc01 = __builtin_amdgcn_wmma_f32_16x16x32_bf16(false, a0, false, b1, (short)0, acc01, false, false);
    acc10 = __builtin_amdgcn_wmma_f32_16x16x32_bf16(false, a1, false, b0, (short)0, acc10, false, false);
    acc11 = __builtin_amdgcn_wmma_f32_16x16x32_bf16(false, a1, false, b1, (short)0, acc11, false, false);
  }

  const int   nlo = n0 + l15, nhi = nlo + 16;
  const float blo = bias ? bias[nlo] : 0.0f;
  const float bhi = bias ? bias[nhi] : 0.0f;
#pragma unroll
  for (int r = 0; r < 8; ++r) {
    const int mA = m0 + r + half * 8;          // D: VGPR r, lanes 0-15 -> M=r
    const int mB = mA + 16;
    const float v00 = acc00[r] + blo, v01 = acc01[r] + bhi;
    const float v10 = acc10[r] + blo, v11 = acc11[r] + bhi;
    C[(size_t)mA * ldc + nlo] = v00;
    C[(size_t)mA * ldc + nhi] = v01;
    C[(size_t)mB * ldc + nlo] = v10;
    C[(size_t)mB * ldc + nhi] = v11;
    if (Cb) {
      Cb[(size_t)mA * ldcb + nlo] = (__bf16)v00;
      Cb[(size_t)mA * ldcb + nhi] = (__bf16)v01;
      Cb[(size_t)mB * ldcb + nlo] = (__bf16)v10;
      Cb[(size_t)mB * ldcb + nhi] = (__bf16)v11;
    }
  }
}

// ---------------------------------------------------------------------------
// Fused stripe kernel:  h = LN(h + A @ W^T + gbias) * ls + lb   (N == E_ == 512)
// One block owns 16 rows.  512 threads = 16 waves; wave w computes the 16x32
// tile at n0 = 32*w (2 WMMA chains).  A stripe staged once in LDS via the
// CDNA5 async global->LDS DMA path (ASYNCcnt), GEMM result staged in LDS,
// then wave-per-row LayerNorm with wave32 shuffle reductions.
// Updates f32 master + bf16 mirror of h.
// ---------------------------------------------------------------------------
#define SAP 520    // LDS row stride for bf16 A stripe (512 + 8 pad)
#define SCP 516    // LDS row stride for f32 C stripe  (512 + 4 pad)

__global__ __launch_bounds__(512) void gemm_ln_fused(
    const __bf16* __restrict__ A,        // [ROWS, E_]
    const __bf16* __restrict__ W,        // [E_, E_]
    const float*  __restrict__ gbias,    // [E_]
    float*        __restrict__ hf,       // [ROWS, E_] residual in / LN out
    __bf16*       __restrict__ hb,       // bf16 mirror of hf
    const float*  __restrict__ ls,
    const float*  __restrict__ lb)
{
  __shared__ alignas(32) __bf16 sA[16 * SAP];
  __shared__ alignas(16) float  sC[16 * SCP];

  const int tid  = threadIdx.x;
  const int lane = tid & 31;
  const int wave = tid >> 5;             // 0..15
  const int row0 = blockIdx.x * 16;

  // Stage the 16x512 bf16 A stripe with async global->LDS DMA:
  // 32 threads per row, 2 x b128 transfers each (no VGPR data, ASYNCcnt).
  {
    const int r = tid >> 5;              // row 0..15
    const int c = tid & 31;              // chunk lane
    const __bf16* src = A + (size_t)(row0 + r) * E_;
    __bf16*       dst = sA + r * SAP;
#pragma unroll
    for (int j = 0; j < 2; ++j) {
      const int ch = c + 32 * j;         // 0..63 chunks of 8 bf16 (16B)
      // Low 32 bits of a generic pointer to LDS == LDS byte address (ISA 10.2).
      const uint32_t      ldsa = (uint32_t)(uintptr_t)(dst + ch * 8);
      const __bf16* const ga   = src + ch * 8;
      asm volatile("global_load_async_to_lds_b128 %0, %1, off"
                   :: "v"(ldsa), "v"(ga) : "memory");
    }
    asm volatile("s_wait_asynccnt 0" ::: "memory");
  }
  __syncthreads();

  const int half = lane >> 4;
  const int l15  = lane & 15;
  const int n0   = wave * 32;

  const __bf16* arow  = sA + l15 * SAP;
  const __bf16* wrow0 = W + (size_t)(n0 + l15) * E_;
  const __bf16* wrow1 = wrow0 + (size_t)16 * E_;

  v8f acc0 = {}, acc1 = {};
  for (int k0 = 0; k0 < E_; k0 += 32) {
    __builtin_prefetch(wrow0 + k0 + 64, 0, 1);
    v16bf a  = load_a_frag(arow, k0, half);          // ds path
    v16bf b0 = *(const v16bf*)(wrow0 + k0 + half * 16);
    v16bf b1 = *(const v16bf*)(wrow1 + k0 + half * 16);
    acc0 = __builtin_amdgcn_wmma_f32_16x16x32_bf16(false, a, false, b0, (short)0, acc0, false, false);
    acc1 = __builtin_amdgcn_wmma_f32_16x16x32_bf16(false, a, false, b1, (short)0, acc1, false, false);
  }

  // GEMM epilogue -> LDS (add gemm bias here)
  {
    const int nlo = n0 + l15, nhi = nlo + 16;
    const float blo = gbias[nlo], bhi = gbias[nhi];
#pragma unroll
    for (int r = 0; r < 8; ++r) {
      const int m = r + half * 8;
      sC[m * SCP + nlo] = acc0[r] + blo;
      sC[m * SCP + nhi] = acc1[r] + bhi;
    }
  }
  __syncthreads();

  // LayerNorm: wave w handles row w of the stripe (512 elems, 16 per lane).
  {
    const int row = row0 + wave;
    const float* hrow = hf + (size_t)row * E_;
    const float* crow = sC + wave * SCP;
    float vals[16];
    float sum = 0.f;
#pragma unroll
    for (int i = 0; i < 16; ++i) {
      const int e = lane + i * 32;
      const float v = hrow[e] + crow[e];
      vals[i] = v; sum += v;
    }
    for (int off = 16; off; off >>= 1) sum += __shfl_xor(sum, off, 32);
    const float mu = sum * (1.f / 512.f);
    float var = 0.f;
#pragma unroll
    for (int i = 0; i < 16; ++i) { const float d = vals[i] - mu; var += d * d; }
    for (int off = 16; off; off >>= 1) var += __shfl_xor(var, off, 32);
    const float inv = rsqrtf(var * (1.f / 512.f) + 1e-5f);

    float*  ho  = hf + (size_t)row * E_;
    __bf16* hbo = hb + (size_t)row * E_;
#pragma unroll
    for (int i = 0; i < 16; ++i) {
      const int e = lane + i * 32;
      const float y = (vals[i] - mu) * inv * ls[e] + lb[e];
      ho[e] = y; hbo[e] = (__bf16)y;
    }
  }
}

// ---------------------------------------------------------------------------
// Attention for one (batch, head): 16x16 scores in f32, mask, softmax, AV.
// qkv layout: row = b*16 + pos, 1536 cols = [q | k | v], head h at h*64.
// ---------------------------------------------------------------------------
__global__ __launch_bounds__(256) void attn_kernel(
    const float* __restrict__ qkv,
    __bf16*      __restrict__ ao,    // [ROWS, E_] bf16 (pre out-proj)
    int t)
{
  const int bh = blockIdx.x;
  const int b  = bh >> 3;
  const int h  = bh & 7;
  const int tid = threadIdx.x;
  const int q = tid >> 4, k = tid & 15;

  __shared__ float sc[16][17];
  __shared__ float at[16][17];

  const float* qrow = qkv + (size_t)(b * 16 + q) * 1536 + h * 64;
  const float* krow = qkv + (size_t)(b * 16 + k) * 1536 + 512 + h * 64;
  float s = 0.f;
#pragma unroll
  for (int d = 0; d < 64; ++d) s += qrow[d] * krow[d];
  s *= 0.125f;                                   // 1/sqrt(64)
  const bool masked = ((q > t) || (k > t)) && (k != 0);
  sc[q][k] = masked ? -1.0e9f : s;
  __syncthreads();

  if (tid < 16) {
    float mx = sc[tid][0];
#pragma unroll
    for (int j = 1; j < 16; ++j) mx = fmaxf(mx, sc[tid][j]);
    float sum = 0.f;
#pragma unroll
    for (int j = 0; j < 16; ++j) { float e = __expf(sc[tid][j] - mx); at[tid][j] = e; sum += e; }
    const float inv = 1.f / sum;
#pragma unroll
    for (int j = 0; j < 16; ++j) at[tid][j] *= inv;
  }
  __syncthreads();

  const int d0 = (tid & 15) * 4;                 // 16x64 outputs / 256 threads
  float o0 = 0.f, o1 = 0.f, o2 = 0.f, o3 = 0.f;
#pragma unroll
  for (int kk = 0; kk < 16; ++kk) {
    const float a = at[q][kk];
    const float* vrow = qkv + (size_t)(b * 16 + kk) * 1536 + 1024 + h * 64 + d0;
    o0 += a * vrow[0]; o1 += a * vrow[1]; o2 += a * vrow[2]; o3 += a * vrow[3];
  }
  __bf16* orow = ao + (size_t)(b * 16 + q) * E_ + h * 64 + d0;
  orow[0] = (__bf16)o0; orow[1] = (__bf16)o1; orow[2] = (__bf16)o2; orow[3] = (__bf16)o3;
}

// ---------------------------------------------------------------------------
// Utility kernels
// ---------------------------------------------------------------------------
__global__ void f32_to_bf16(const float* __restrict__ in, __bf16* __restrict__ out, int n) {
  const int i = blockIdx.x * 256 + threadIdx.x;
  if (i < n) out[i] = (__bf16)in[i];
}

__global__ void init_h_kernel(const float* __restrict__ h0,
                              float* __restrict__ hf, __bf16* __restrict__ hb) {
  const int i = blockIdx.x * 256 + threadIdx.x;     // over B_*S_*E_ = 1M
  const int e = i % E_;
  const int p = (i / E_) % S_;
  const int b = i / (E_ * S_);
  const float v = h0[(size_t)b * MEM_ + p * E_ + e];
  hf[i] = v; hb[i] = (__bf16)v;
}

__global__ void write_newh_kernel(const float* __restrict__ hf,
                                  const float* __restrict__ h0,
                                  float* __restrict__ out) {
  const int i = blockIdx.x * 256 + threadIdx.x;
  if (i >= B_ * MEM_) return;
  const int b = i / MEM_;
  const int r = i % MEM_;
  float v;
  if (r == MEM_ - 1) {
    v = h0[(size_t)b * MEM_ + (MEM_ - 1)] + (float)S_;   // ci_fin
  } else {
    const int p = r / E_, e = r % E_;
    v = (p < S_) ? hf[((size_t)b * S_ + p) * E_ + e]
                 : h0[(size_t)b * MEM_ + r];             // untouched pad region
  }
  out[i] = v;
}

// ---------------------------------------------------------------------------
// Host orchestration
// ---------------------------------------------------------------------------
extern "C" void kernel_launch(void* const* d_in, const int* in_sizes, int n_in,
                              void* d_out, int out_size, void* d_ws, size_t ws_size,
                              hipStream_t stream) {
  const float* seq        = (const float*)d_in[0];   // [S,B,IN]
  const float* h_state    = (const float*)d_in[1];   // [1,B,MEM]
  const float* Win        = (const float*)d_in[2];   // [E,IN]
  const float* b_in       = (const float*)d_in[3];   // [E]
  const float* Wout       = (const float*)d_in[4];   // [OUT,E]
  const float* b_out      = (const float*)d_in[5];   // [OUT]
  const float* in_proj_w  = (const float*)d_in[6];   // [NL,3E,E]
  const float* in_proj_b  = (const float*)d_in[7];   // [NL,3E]
  const float* attn_out_w = (const float*)d_in[8];   // [NL,E,E]
  const float* attn_out_b = (const float*)d_in[9];   // [NL,E]
  const float* ln1_s      = (const float*)d_in[10];
  const float* ln1_b      = (const float*)d_in[11];
  const float* ln2_s      = (const float*)d_in[12];
  const float* ln2_b      = (const float*)d_in[13];
  const float* ff_w       = (const float*)d_in[14];  // [NL,E,E]
  const float* ff_b       = (const float*)d_in[15];  // [NL,E]
  float* out = (float*)d_out;

  char* ws = (char*)d_ws;
  size_t off = 0;
  auto alloc = [&](size_t bytes) -> void* {
    void* p = ws + off;
    off = (off + bytes + 255) & ~(size_t)255;
    return p;
  };
  float*  h_f32   = (float*) alloc((size_t)ROWS * E_ * 4);
  float*  qkv_f32 = (float*) alloc((size_t)ROWS * 3 * E_ * 4);
  __bf16* h_b16   = (__bf16*)alloc((size_t)ROWS * E_ * 2);
  __bf16* ao_b16  = (__bf16*)alloc((size_t)ROWS * E_ * 2);
  __bf16* seq_b16 = (__bf16*)alloc((size_t)S_ * B_ * IN_ * 2);
  __bf16* Win_b   = (__bf16*)alloc((size_t)E_ * IN_ * 2);
  __bf16* ipw_b   = (__bf16*)alloc((size_t)NL_ * 3 * E_ * E_ * 2);
  __bf16* aow_b   = (__bf16*)alloc((size_t)NL_ * E_ * E_ * 2);
  __bf16* ffw_b   = (__bf16*)alloc((size_t)NL_ * E_ * E_ * 2);
  __bf16* Wout_b  = (__bf16*)alloc((size_t)OUT_ * E_ * 2);

  auto cvt = [&](const float* src, __bf16* dst, int n) {
    f32_to_bf16<<<(n + 255) / 256, 256, 0, stream>>>(src, dst, n);
  };
  cvt(seq,        seq_b16, S_ * B_ * IN_);
  cvt(Win,        Win_b,   E_ * IN_);
  cvt(in_proj_w,  ipw_b,   NL_ * 3 * E_ * E_);
  cvt(attn_out_w, aow_b,   NL_ * E_ * E_);
  cvt(ff_w,       ffw_b,   NL_ * E_ * E_);
  cvt(Wout,       Wout_b,  OUT_ * E_);

  init_h_kernel<<<(B_ * S_ * E_) / 256, 256, 0, stream>>>(h_state, h_f32, h_b16);

  for (int t = 0; t < S_; ++t) {
    // x_t = seq[t] @ Win^T + b_in  -> scatter into slot t of h (f32 + bf16)
    gemm_bf16_wmma<<<dim3(E_ / 128, B_ / 64), 256, 0, stream>>>(
        seq_b16 + (size_t)t * B_ * IN_, IN_, Win_b, IN_, b_in,
        h_f32 + (size_t)t * E_, S_ * E_, h_b16 + (size_t)t * E_, S_ * E_, IN_);

    for (int l = 0; l < NL_; ++l) {
      // QKV projection: [2048,512] @ [512,1536]
      gemm_bf16_wmma<<<dim3(3 * E_ / 128, ROWS / 64), 256, 0, stream>>>(
          h_b16, E_, ipw_b + (size_t)l * 3 * E_ * E_, E_,
          in_proj_b + (size_t)l * 3 * E_, qkv_f32, 3 * E_,
          (__bf16*)nullptr, 0, E_);

      attn_kernel<<<B_ * H_, 256, 0, stream>>>(qkv_f32, ao_b16, t);

      // attn out-proj + residual + LN1 (fused)
      gemm_ln_fused<<<ROWS / 16, 512, 0, stream>>>(
          ao_b16, aow_b + (size_t)l * E_ * E_, attn_out_b + (size_t)l * E_,
          h_f32, h_b16, ln1_s + (size_t)l * E_, ln1_b + (size_t)l * E_);

      // feed-forward + residual + LN2 (fused)
      gemm_ln_fused<<<ROWS / 16, 512, 0, stream>>>(
          h_b16, ffw_b + (size_t)l * E_ * E_, ff_b + (size_t)l * E_,
          h_f32, h_b16, ln2_s + (size_t)l * E_, ln2_b + (size_t)l * E_);
    }

    // outputs[t] = h[:, t] @ Wout^T + b_out  (rows strided by S_*E_)
    gemm_bf16_wmma<<<dim3(OUT_ / 128, B_ / 64), 256, 0, stream>>>(
        h_b16 + (size_t)t * E_, S_ * E_, Wout_b, E_, b_out,
        out + (size_t)t * B_ * OUT_, OUT_, (__bf16*)nullptr, 0, E_);
  }

  write_newh_kernel<<<(B_ * MEM_ + 255) / 256, 256, 0, stream>>>(
      h_f32, h_state, out + (size_t)S_ * B_ * OUT_);
}
